// MultiHeadAttention_15470472200797
// MI455X (gfx1250) — matrix operand order repaired and checked by reading, not compiled, not run on hardware
//
#include <hip/hip_runtime.h>
#include <hip/hip_bf16.h>
#include <stdint.h>

#define HEAD 16
#define EMD  1024
#define QK   64
#define VDIM 64
#define BATCH 4
#define SQ   2048
#define SKV  2048

typedef __attribute__((ext_vector_type(16))) __bf16   bf16x16;
typedef __attribute__((ext_vector_type(8)))  float    f32x8;
typedef __attribute__((ext_vector_type(4)))  uint32_t u32x4;

union FragAB { u32x4 q[2]; bf16x16 v; };
union Chunk8 { u32x4 q; uint16_t s[8]; };

__device__ __forceinline__ uint16_t f32_to_bf16(float f) {
  uint32_t u = __float_as_uint(f);
  uint32_t r = u + 0x7FFFu + ((u >> 16) & 1u);     // round-to-nearest-even
  if ((u & 0x7F800000u) == 0x7F800000u) r = u;     // inf/nan passthrough
  return (uint16_t)(r >> 16);
}

// CDNA5 async global->LDS copy (ASYNCcnt). INST_OFFSET applies to BOTH the LDS
// and global addresses (ISA §15.18), so one base + immediate offsets covers a
// 64B stream with a single 64-bit address computation.
#define ASYNC_LDS_B128(lds_off, gaddr, OFF)                              \
  asm volatile("global_load_async_to_lds_b128 %0, %1, off offset:" #OFF \
               :: "v"(lds_off), "v"(gaddr) : "memory")

__device__ __forceinline__ void async_copy64(uint32_t lds_off, const void* gptr) {
  const uint64_t g = (uint64_t)(uintptr_t)gptr;
  ASYNC_LDS_B128(lds_off, g, 0);
  ASYNC_LDS_B128(lds_off, g, 16);
  ASYNC_LDS_B128(lds_off, g, 32);
  ASYNC_LDS_B128(lds_off, g, 48);
}
__device__ __forceinline__ void wait_async0() {
  asm volatile("s_wait_asynccnt 0x0" ::: "memory");
}

// ---------------------------------------------------------------- convert ---
__global__ void cvt_f32_bf16(const float* __restrict__ src,
                             uint16_t* __restrict__ dst, int n) {
  int i = blockIdx.x * blockDim.x + threadIdx.x;
  int stride = gridDim.x * blockDim.x;
  for (; i < n; i += stride) dst[i] = f32_to_bf16(src[i]);
}

// -------------------------------------------------------------- QKV GEMM ----
// Y[m,n] = sum_k A[m,k] * W[n,k] + bias[n]; output scattered to [B,H,S,64] bf16.
// Ping-pong double-buffered LDS: tile k+1 streams in via async DMA while WMMAs
// consume tile k.
#define GM_TILE 128
#define GN_TILE 128
#define GK_TILE 64
#define GA_PITCH (GK_TILE * 2 + 16)  // 144 B: 16-byte pad vs LDS bank conflicts
#define GB_PITCH (GK_TILE * 2 + 16)
#define GSTAGE   (GM_TILE * GA_PITCH + GN_TILE * GB_PITCH)  // 36864 B / stage

__global__ __launch_bounds__(256)
void qkv_gemm(const uint16_t* __restrict__ A,     // [8192,1024] bf16
              const uint16_t* __restrict__ W,     // [1024,1024] bf16 (row n = out feature)
              const float*    __restrict__ bias,  // [1024]
              uint16_t*       __restrict__ out)   // [B,H,S,64] bf16
{
  __shared__ unsigned char smem[2 * GSTAGE];
  const uint32_t sm_off = (uint32_t)(uintptr_t)smem;

  const int tid  = threadIdx.x;
  const int lane = tid & 31;
  const int wave = tid >> 5;
  const int hlf  = lane >> 4;     // wave32 half
  const int r16  = lane & 15;
  const int wm   = wave >> 1;     // 0..3 -> 32-row sub-tile
  const int wn   = wave & 1;      // 0..1 -> 64-col sub-tile
  const int mbase = blockIdx.x * GM_TILE;
  const int nbase = blockIdx.y * GN_TILE;

  // staging geometry: 128 rows x 64 k (128B/row); 256 threads x 64B each
  const int rowS = tid >> 1;
  const int cbS  = (tid & 1) * 64;   // byte offset within the 128B row
  const uint16_t* gA = A + (size_t)(mbase + rowS) * EMD + cbS / 2;
  const uint16_t* gW = W + (size_t)(nbase + rowS) * EMD + cbS / 2;
  const uint32_t  lA = sm_off + rowS * GA_PITCH + cbS;
  const uint32_t  lB = sm_off + GM_TILE * GA_PITCH + rowS * GB_PITCH + cbS;

  f32x8 acc[2][4] = {};

  // prologue: stage k-tile 0 into buffer 0
  async_copy64(lA, gA);
  async_copy64(lB, gW);

  int cur = 0;
  for (int k0 = 0; k0 < EMD; k0 += GK_TILE) {
    wait_async0();                 // my DMA for buf[cur] complete
    __syncthreads();               // everyone's buf[cur] published
    if (k0 + GK_TILE < EMD) {      // stream next tile into the other buffer
      async_copy64(lA + (cur ^ 1) * GSTAGE, gA + k0 + GK_TILE);
      async_copy64(lB + (cur ^ 1) * GSTAGE, gW + k0 + GK_TILE);
      if (k0 + 2 * GK_TILE < EMD) {
        __builtin_prefetch(gA + k0 + 2 * GK_TILE, 0, 0);  // global_prefetch_b8
        __builtin_prefetch(gW + k0 + 2 * GK_TILE, 0, 0);
      }
    }

    const unsigned char* smA = smem + cur * GSTAGE;
    const unsigned char* smB = smA + GM_TILE * GA_PITCH;
#pragma unroll
    for (int ks = 0; ks < 2; ++ks) {          // two 32-wide WMMA K-steps
      const int kb = ks * 64;                 // byte offset within LDS row
      FragAB a[2], bfr[4];
#pragma unroll
      for (int t = 0; t < 2; ++t) {
        // A frag (16x32): lanes 0-15 K{0..7,16..23}; lanes 16-31 K{8..15,24..31}
        const unsigned char* pa = smA + (wm * 32 + t * 16 + r16) * GA_PITCH + kb + hlf * 16;
        a[t].q[0] = *(const u32x4*)(pa);
        a[t].q[1] = *(const u32x4*)(pa + 32);
      }
#pragma unroll
      for (int t = 0; t < 4; ++t) {
        // B frag (32x16): lanes 0-15 K{0..7,16..23}; lanes 16-31 K{8..15,24..31} of column n
        const unsigned char* pb = smB + (wn * 64 + t * 16 + r16) * GB_PITCH + kb + hlf * 16;
        bfr[t].q[0] = *(const u32x4*)(pb);
        bfr[t].q[1] = *(const u32x4*)(pb + 32);
      }
#pragma unroll
      for (int i = 0; i < 2; ++i)
#pragma unroll
        for (int j = 0; j < 4; ++j)
          acc[i][j] = __builtin_amdgcn_wmma_f32_16x16x32_bf16(
              false, a[i].v, false, bfr[j].v, (short)0, acc[i][j], false, false);
    }
    cur ^= 1;
  }

  // Epilogue: + bias, convert to bf16, scatter into [B,H,S,64]
#pragma unroll
  for (int i = 0; i < 2; ++i) {
#pragma unroll
    for (int j = 0; j < 4; ++j) {
      const int n = nbase + wn * 64 + j * 16 + r16;
      const int h = n >> 6, d = n & 63;
      const float bv = bias[n];
#pragma unroll
      for (int e = 0; e < 8; ++e) {
        const int m  = mbase + wm * 32 + i * 16 + e + hlf * 8;  // C layout: M = e (+8 upper half)
        const int bb = m >> 11;
        const int s  = m & 2047;
        out[(((size_t)(bb * HEAD + h) * SQ + s) << 6) + d] = f32_to_bf16(acc[i][j][e] + bv);
      }
    }
  }
}

// ---------------------------------------------------------- flash attention --
// Double-buffered K (async DMA) and Vt (register prefetch + transposed scatter).
#define KV_TILE  64
#define KT_PITCH 144
#define VT_PITCH 144
#define P_PITCH  144
#define KSTAGE   (KV_TILE * KT_PITCH)   // 9216 B
#define VSTAGE   (KV_TILE * VT_PITCH)   // 9216 B

__global__ __launch_bounds__(128)
void flash_attn(const uint16_t* __restrict__ Q,   // [B,H,SQ,64]  bf16
                const uint16_t* __restrict__ K,   // [B,H,SKV,64] bf16
                const uint16_t* __restrict__ V,   // [B,H,SKV,64] bf16
                const unsigned char* __restrict__ mask, // [B,SQ,SKV] bool (1 byte)
                float* __restrict__ out)          // [B,SQ,H*64] fp32
{
  __shared__ unsigned char smem[2 * KSTAGE + 2 * VSTAGE + 4 * 16 * P_PITCH];
  unsigned char* smK  = smem;                      // two K stages
  unsigned char* smVt = smem + 2 * KSTAGE;         // two Vt stages
  unsigned char* smP  = smVt + 2 * VSTAGE;
  const uint32_t smK_off = (uint32_t)(uintptr_t)smK;

  const int tid  = threadIdx.x;
  const int lane = tid & 31;
  const int wave = tid >> 5;    // 0..3, each owns a 16-row q strip
  const int hlf  = lane >> 4;
  const int r16  = lane & 15;

  const int qt = blockIdx.x;    // 64-row q tile
  const int h  = blockIdx.y;
  const int b  = blockIdx.z;

  const int qbase = qt * 64 + wave * 16;
  const uint16_t* Qp = Q + ((size_t)(b * HEAD + h) * SQ)  * 64;
  const uint16_t* Kp = K + ((size_t)(b * HEAD + h) * SKV) * 64;
  const uint16_t* Vp = V + ((size_t)(b * HEAD + h) * SKV) * 64;
  const unsigned char* Mp = mask + (size_t)b * SQ * SKV;

  // staging geometry: 64 rows x 128B; 128 threads x 64B each
  const int rowS = tid >> 1;
  const int cbS  = (tid & 1) * 64;
  const uint16_t* gK = Kp + (size_t)rowS * 64 + cbS / 2;
  const uint16_t* gV = Vp + (size_t)rowS * 64 + cbS / 2;
  const uint32_t  lK = smK_off + rowS * KT_PITCH + cbS;

  // Q fragments (K-dim 64 -> two 16x32 A frags) held in registers for the kernel.
  FragAB qf[2];
  {
    const uint16_t* qrow = Qp + (size_t)(qbase + r16) * 64;
#pragma unroll
    for (int kh = 0; kh < 2; ++kh) {
      qf[kh].q[0] = *(const u32x4*)(qrow + kh * 32 + hlf * 8);
      qf[kh].q[1] = *(const u32x4*)(qrow + kh * 32 + 16 + hlf * 8);
    }
  }

  f32x8 oacc[4] = {};
  float m_i[8], l_i[8];
#pragma unroll
  for (int e = 0; e < 8; ++e) { m_i[e] = -3.0e38f; l_i[e] = 0.0f; }
  const float scale = 0.125f;   // 1/sqrt(QK=64)

  // prologue: stage kv-tile 0 (K via async DMA, V into registers)
  u32x4 vreg[4];
  async_copy64(lK, gK);
#pragma unroll
  for (int c = 0; c < 4; ++c) vreg[c] = ((const u32x4*)gV)[c];

  int cur = 0;
  for (int kv0 = 0; kv0 < SKV; kv0 += KV_TILE) {
    wait_async0();                          // K[cur] DMA complete (this wave)
    // transpose-scatter prefetched V registers into Vt[cur]
    {
      unsigned char* vt = smVt + cur * VSTAGE;
#pragma unroll
      for (int c = 0; c < 4; ++c) {
        Chunk8 ch; ch.q = vreg[c];
        const int d0 = cbS / 2 + c * 8;
#pragma unroll
        for (int e = 0; e < 8; ++e)
          *(uint16_t*)(vt + (d0 + e) * VT_PITCH + rowS * 2) = ch.s[e];
      }
    }
    __syncthreads();                        // K[cur] + Vt[cur] published

    if (kv0 + KV_TILE < SKV) {              // stream next tile into other buffers
      async_copy64(lK + (cur ^ 1) * KSTAGE, gK + (kv0 + KV_TILE) * 64);
      const u32x4* nv = (const u32x4*)(gV + (kv0 + KV_TILE) * 64);
#pragma unroll
      for (int c = 0; c < 4; ++c) vreg[c] = nv[c];
    }

    const unsigned char* kb = smK + cur * KSTAGE;
    const unsigned char* vt = smVt + cur * VSTAGE;

    // S = (Q K^T) * scale, with padding mask
    f32x8 sacc[4];
#pragma unroll
    for (int nt = 0; nt < 4; ++nt) {
      f32x8 c = {};
#pragma unroll
      for (int kh = 0; kh < 2; ++kh) {
        FragAB bfr;
        const unsigned char* pb = kb + (nt * 16 + r16) * KT_PITCH + kh * 64 + hlf * 16;
        bfr.q[0] = *(const u32x4*)(pb);
        bfr.q[1] = *(const u32x4*)(pb + 32);
        c = __builtin_amdgcn_wmma_f32_16x16x32_bf16(
            false, qf[kh].v, false, bfr.v, (short)0, c, false, false);
      }
      const int kvcol = kv0 + nt * 16 + r16;
#pragma unroll
      for (int e = 0; e < 8; ++e) {
        const int qrow = qbase + e + hlf * 8;
        c[e] = Mp[(size_t)qrow * SKV + kvcol] ? -1.0e9f : (c[e] * scale);
      }
      sacc[nt] = c;
    }

    // online softmax: row max across 4 tiles + 16 lanes of the half-wave
    float mnew[8], alpha[8];
#pragma unroll
    for (int e = 0; e < 8; ++e) {
      float v = fmaxf(fmaxf(sacc[0][e], sacc[1][e]), fmaxf(sacc[2][e], sacc[3][e]));
#pragma unroll
      for (int m = 1; m < 16; m <<= 1)
        v = fmaxf(v, __shfl_xor(v, m, 32));   // xor<16 stays within the half
      mnew[e]  = fmaxf(m_i[e], v);
      alpha[e] = __expf(m_i[e] - mnew[e]);
      m_i[e]   = mnew[e];
    }

    // P = exp(S - mnew); write bf16 P to this wave's LDS strip (A-major rows)
    unsigned char* pw = smP + wave * 16 * P_PITCH;
#pragma unroll
    for (int nt = 0; nt < 4; ++nt)
#pragma unroll
      for (int e = 0; e < 8; ++e) {
        float p = __expf(sacc[nt][e] - mnew[e]);
        sacc[nt][e] = p;
        *(uint16_t*)(pw + (e + hlf * 8) * P_PITCH + (nt * 16 + r16) * 2) = f32_to_bf16(p);
      }

    // running denominator + rescale O
#pragma unroll
    for (int e = 0; e < 8; ++e) {
      float rs = sacc[0][e] + sacc[1][e] + sacc[2][e] + sacc[3][e];
#pragma unroll
      for (int m = 1; m < 16; m <<= 1)
        rs += __shfl_xor(rs, m, 32);
      l_i[e] = l_i[e] * alpha[e] + rs;
    }
#pragma unroll
    for (int nt = 0; nt < 4; ++nt)
#pragma unroll
      for (int e = 0; e < 8; ++e)
        oacc[nt][e] *= alpha[e];

    // O += P(16x64) * V(64x64) — P frags loaded ONCE, reused across 4 v-tiles
    FragAB pa[2];
#pragma unroll
    for (int kh = 0; kh < 2; ++kh) {
      const unsigned char* pp = pw + r16 * P_PITCH + kh * 64 + hlf * 16;
      pa[kh].q[0] = *(const u32x4*)(pp);
      pa[kh].q[1] = *(const u32x4*)(pp + 32);
    }
#pragma unroll
    for (int vtile = 0; vtile < 4; ++vtile) {
#pragma unroll
      for (int kh = 0; kh < 2; ++kh) {
        FragAB vb;
        const unsigned char* pv = vt + (vtile * 16 + r16) * VT_PITCH + kh * 64 + hlf * 16;
        vb.q[0] = *(const u32x4*)(pv);
        vb.q[1] = *(const u32x4*)(pv + 32);
        oacc[vtile] = __builtin_amdgcn_wmma_f32_16x16x32_bf16(
            false, pa[kh].v, false, vb.v, (short)0, oacc[vtile], false, false);
      }
    }
    cur ^= 1;
  }

  // epilogue: O / l, write [B,SQ,H*64] fp32
#pragma unroll
  for (int vtile = 0; vtile < 4; ++vtile) {
    const int d = vtile * 16 + r16;
#pragma unroll
    for (int e = 0; e < 8; ++e) {
      const int qrow = qbase + e + hlf * 8;
      out[(size_t)(b * SQ + qrow) * (HEAD * VDIM) + h * VDIM + d] = oacc[vtile][e] / l_i[e];
    }
  }
}

// ----------------------------------------------------------------- launch ---
extern "C" void kernel_launch(void* const* d_in, const int* in_sizes, int n_in,
                              void* d_out, int out_size, void* d_ws, size_t ws_size,
                              hipStream_t stream) {
  const float* x_q  = (const float*)d_in[0];
  const float* x_kv = (const float*)d_in[1];
  const unsigned char* mask = (const unsigned char*)d_in[2]; // bool array (1B/elem)
  const float* Wq   = (const float*)d_in[3];
  const float* Wq_b = (const float*)d_in[4];
  const float* Wk   = (const float*)d_in[5];
  const float* Wk_b = (const float*)d_in[6];
  const float* Wv   = (const float*)d_in[7];
  const float* Wv_b = (const float*)d_in[8];
  float* out = (float*)d_out;

  unsigned char* ws = (unsigned char*)d_ws;
  const size_t MB = 1024u * 1024u;
  uint16_t* xq_bf  = (uint16_t*)(ws + 0 * MB);   // 16 MB
  uint16_t* xkv_bf = (uint16_t*)(ws + 16 * MB);  // 16 MB
  uint16_t* wq_bf  = (uint16_t*)(ws + 32 * MB);  //  2 MB
  uint16_t* wk_bf  = (uint16_t*)(ws + 34 * MB);  //  2 MB
  uint16_t* wv_bf  = (uint16_t*)(ws + 36 * MB);  //  2 MB
  uint16_t* Qb     = (uint16_t*)(ws + 38 * MB);  // 16 MB  [B,H,SQ,64]
  uint16_t* Kb     = (uint16_t*)(ws + 54 * MB);  // 16 MB
  uint16_t* Vb     = (uint16_t*)(ws + 70 * MB);  // 16 MB

  const int nX = BATCH * SQ * EMD;   // 8,388,608
  const int nW = EMD * EMD;          // 1,048,576
  cvt_f32_bf16<<<dim3(2048), dim3(256), 0, stream>>>(x_q,  xq_bf,  nX);
  cvt_f32_bf16<<<dim3(2048), dim3(256), 0, stream>>>(x_kv, xkv_bf, nX);
  cvt_f32_bf16<<<dim3(1024), dim3(256), 0, stream>>>(Wq, wq_bf, nW);
  cvt_f32_bf16<<<dim3(1024), dim3(256), 0, stream>>>(Wk, wk_bf, nW);
  cvt_f32_bf16<<<dim3(1024), dim3(256), 0, stream>>>(Wv, wv_bf, nW);

  dim3 gg((BATCH * SQ) / GM_TILE, EMD / GN_TILE);  // 64 x 8
  qkv_gemm<<<gg, 256, 0, stream>>>(xq_bf,  wq_bf, Wq_b, Qb);
  qkv_gemm<<<gg, 256, 0, stream>>>(xkv_bf, wk_bf, Wk_b, Kb);
  qkv_gemm<<<gg, 256, 0, stream>>>(xkv_bf, wv_bf, Wv_b, Vb);

  flash_attn<<<dim3(SQ / 64, HEAD, BATCH), 128, 0, stream>>>(Qb, Kb, Vb, mask, out);
}
